// GraphAttentionLayer_82540681494981
// MI455X (gfx1250) — matrix operand order repaired and checked by reading, not compiled, not run on hardware
//
#include <hip/hip_runtime.h>
#include <hip/hip_bf16.h>

// GAT layer, MI455X (gfx1250, wave32, WMMA).
// B=4, N=2048, FIN=512, FOUT=128. HBM-bound on adj (67MB) -> single fused pass,
// transposed-value formulation: D^T = V^T x P^T so softmax state is per-lane.

typedef __attribute__((ext_vector_type(16))) _Float16 v16h;
typedef __attribute__((ext_vector_type(8)))  _Float16 v8h;
typedef __attribute__((ext_vector_type(8)))  float    v8f;
typedef __attribute__((ext_vector_type(4)))  float    v4f;
typedef __attribute__((ext_vector_type(4)))  int      v4i;

#define B_     4
#define N_     2048
#define FIN_   512
#define FOUT_  128
#define NEG_INF_ (-9.0e15f)
#define ALPHA_   0.2f

union AFrag { v16h v; v8h h[2]; _Float16 e[16]; };
union CFrag { v8f  v; float e[8]; };

// ---------------------------------------------------------------------------
// K0: Wt[f][k] = (f16) W[k][f]   (128KB, built once, lives in L2)
// ---------------------------------------------------------------------------
__global__ __launch_bounds__(256)
void gat_wt(const float* __restrict__ W, _Float16* __restrict__ Wt) {
  int o = blockIdx.x * 256 + threadIdx.x;   // o = f*FIN + k
  int k = o & (FIN_ - 1);
  int f = o >> 9;
  Wt[o] = (_Float16)W[(size_t)k * FOUT_ + f];
}

// ---------------------------------------------------------------------------
// K1: hT[b][f][j] = (x @ W)^T  in f16.
// One wave per 16x16 tile; B-frag = two 16B runs of Wt; epilogue transposes
// the C tile through per-wave LDS (s_wait_dscnt for cross-lane visibility).
// ---------------------------------------------------------------------------
__global__ __launch_bounds__(256)
void gat_gemm_h(const float* __restrict__ x, const _Float16* __restrict__ Wt,
                _Float16* __restrict__ hT) {
  __shared__ float ct[8][16 * 17];          // per-wave padded 16x16 tile

  const int lane = threadIdx.x & 31;
  const int wv   = threadIdx.x >> 5;
  const int mt   = blockIdx.x;              // 16-row tile over B*N
  const int f0   = wv * 16;                 // feature tile (8 waves cover 128)
  const int row  = lane & 15;
  const int kb   = (lane < 16) ? 0 : 8;     // A-frag K half (16-bit A 16x32)
  const int kbb  = (lane < 16) ? 0 : 16;    // unused; B-frag uses kb
  const float* arow = x + (size_t)(mt * 16 + row) * FIN_;
  const _Float16* wrow = Wt + (size_t)(f0 + row) * FIN_;

  CFrag c;
#pragma unroll
  for (int i = 0; i < 8; i++) c.e[i] = 0.f;

  for (int k0 = 0; k0 < FIN_; k0 += 32) {
    AFrag a, b;
    v4f x0 = *(const v4f*)(arow + k0 + kb);
    v4f x1 = *(const v4f*)(arow + k0 + kb + 4);
    v4f x2 = *(const v4f*)(arow + k0 + 16 + kb);
    v4f x3 = *(const v4f*)(arow + k0 + 16 + kb + 4);
#pragma unroll
    for (int i = 0; i < 4; i++) {
      a.e[i]      = (_Float16)x0[i];
      a.e[4 + i]  = (_Float16)x1[i];
      a.e[8 + i]  = (_Float16)x2[i];
      a.e[12 + i] = (_Float16)x3[i];
    }
    b.h[0] = *(const v8h*)(wrow + k0 + kb);       // k order matches A-frag
    b.h[1] = *(const v8h*)(wrow + k0 + 16 + kb);
    c.v = __builtin_amdgcn_wmma_f32_16x16x32_f16(false, a.v, false, b.v,
                                                 (short)0, c.v, false, false);
  }

  // transpose C tile (rows=n, cols=feature) through LDS -> hT rows
#pragma unroll
  for (int v = 0; v < 8; v++) {
    int m = (lane < 16) ? v : v + 8;        // C-frag row mapping
    ct[wv][m * 17 + row] = c.e[v];
  }
  asm volatile("s_wait_dscnt 0x0" ::: "memory");  // cross-lane LDS visibility
  const int half = lane >> 4;               // which 8 n's this lane emits
  const int bb = mt >> 7;                   // batch
  const int nb = (mt & 127) * 16;           // n base within batch
  v8h o_;
#pragma unroll
  for (int u = 0; u < 8; u++)
    o_[u] = (_Float16)ct[wv][(half * 8 + u) * 17 + row];
  *(v8h*)(hT + ((size_t)(bb * FOUT_ + f0 + row)) * N_ + nb + half * 8) = o_;
}

// ---------------------------------------------------------------------------
// K2: t1/t2 per query, coalesced over hT columns (deterministic, no reduce)
// ---------------------------------------------------------------------------
__global__ __launch_bounds__(256)
void gat_tvec(const _Float16* __restrict__ hT, const float* __restrict__ a1,
              const float* __restrict__ a2, float* __restrict__ t1,
              float* __restrict__ t2) {
  const int b = blockIdx.x >> 3;
  const int q = (blockIdx.x & 7) * 256 + threadIdx.x;
  const _Float16* hb = hT + (size_t)b * FOUT_ * N_ + q;
  float s1 = 0.f, s2 = 0.f;
#pragma unroll 8
  for (int f = 0; f < FOUT_; f++) {
    float hv = (float)hb[(size_t)f * N_];
    s1 += hv * a1[f];
    s2 += hv * a2[f];
  }
  t1[(size_t)b * N_ + q] = s1;
  t2[(size_t)b * N_ + q] = s2;
}

// ---------------------------------------------------------------------------
// K3: fused masked-softmax + P@V, transposed form. Block = 8 waves = one
// 16-query tile; each wave streams a 256-wide j segment (4096 waves total to
// keep the 67MB adj stream saturating HBM); deterministic LDS tree merge.
// ---------------------------------------------------------------------------
__global__ __launch_bounds__(256)
void gat_attn(const _Float16* __restrict__ hT, const float* __restrict__ t1,
              const float* __restrict__ t2, const int* __restrict__ adj,
              const int* __restrict__ d1p, const int* __restrict__ d2p,
              float* __restrict__ out) {
  __shared__ float t2s[N_];                 // 8KB
  __shared__ float marr[8][16];             // per-wave row max
  __shared__ float larr[8][16];             // per-wave row sum
  __shared__ float redbuf[4][16 * FOUT_];   // 32KB tree buffer (+ final tile)

  const int t    = threadIdx.x;
  const int lane = t & 31;
  const int wv   = t >> 5;
  const int b    = blockIdx.x >> 7;
  const int i0   = (blockIdx.x & 127) * 16; // query tile base
  const int q    = lane & 15;               // this lane's query (both halves)
  const int kbb  = (lane < 16) ? 0 : 16;    // B-frag (P^T) j half
  const int kb   = (lane < 16) ? 0 : 8;     // unused; A-frag uses kbb
  const int d1 = d1p[0], d2 = d2p[0];
  const size_t bN = (size_t)b * N_;
  const _Float16* hTb = hT + (size_t)b * FOUT_ * N_;

  { // stage t2[b,:]
    const v4f* src = (const v4f*)(t2 + bN);
    v4f* dst = (v4f*)t2s;
    dst[t] = src[t];
    dst[t + 256] = src[t + 256];
  }
  __syncthreads();

  const float t1r = t1[bN + i0 + q];
  const int* adjrow = adj + (bN + i0 + q) * (size_t)N_;

  float m_run = NEG_INF_;                   // matches reference -9e15 fill
  float l_run = 0.f;
  CFrag acc[8];
#pragma unroll
  for (int nt = 0; nt < 8; nt++)
#pragma unroll
    for (int i = 0; i < 8; i++) acc[nt].e[i] = 0.f;

  const int jbeg = wv * (N_ / 8);
  const int jend = jbeg + (N_ / 8);
  for (int j0 = jbeg; j0 < jend; j0 += 32) {
    // adj: 16 contiguous ints per lane (lane halves cover the 32-wide step)
    const v4i* ap = (const v4i*)(adjrow + j0 + kbb);
    v4i a0 = ap[0], a1v = ap[1], a2v = ap[2], a3v = ap[3];
    if (j0 + 32 < jend)                     // gfx1250 global_prefetch_b8
      __builtin_prefetch(adjrow + j0 + 32 + kbb, 0, 3);

    const float* tp = &t2s[j0 + kbb];
    v4f u0 = *(const v4f*)tp;       v4f u1 = *(const v4f*)(tp + 4);
    v4f u2 = *(const v4f*)(tp + 8); v4f u3 = *(const v4f*)(tp + 12);

    float ev[16];
#pragma unroll
    for (int i = 0; i < 16; i++) {
      int av = (i < 4) ? a0[i] : (i < 8) ? a1v[i - 4]
             : (i < 12) ? a2v[i - 8] : a3v[i - 12];
      float tvv = (i < 4) ? u0[i] : (i < 8) ? u1[i - 4]
                : (i < 12) ? u2[i - 8] : u3[i - 12];
      float e = t1r + tvv;
      e = (e > 0.f) ? e : ALPHA_ * e;       // leaky relu
      ev[i] = ((av == d1) || (av == d2)) ? e : NEG_INF_;
    }

    // online softmax; lanes q and q+16 share a query -> combine via xor-16
    float mt_ = ev[0];
#pragma unroll
    for (int i = 1; i < 16; i++) mt_ = fmaxf(mt_, ev[i]);
    mt_ = fmaxf(mt_, __shfl_xor(mt_, 16, 32));
    float m_new = fmaxf(m_run, mt_);
    float scale = __expf(m_run - m_new);    // per-lane constant!
    AFrag p;                                // P^T in B-frag layout
    float ls = 0.f;
#pragma unroll
    for (int i = 0; i < 16; i++) {
      float pv = __expf(ev[i] - m_new);
      ls += pv;
      p.e[i] = (_Float16)pv;
    }
    ls += __shfl_xor(ls, 16, 32);
    l_run = l_run * scale + ls;
    m_run = m_new;

#pragma unroll
    for (int nt = 0; nt < 8; nt++) {
#pragma unroll
      for (int i = 0; i < 8; i++) acc[nt].e[i] *= scale;
      AFrag af;                             // V^T row: contiguous 16B runs
      const _Float16* rp = hTb + (size_t)(nt * 16 + q) * N_;
      af.h[0] = *(const v8h*)(rp + j0 + kbb);
      af.h[1] = *(const v8h*)(rp + j0 + kbb + 8);
      acc[nt].v = __builtin_amdgcn_wmma_f32_16x16x32_f16(
          false, af.v, false, p.v, (short)0, acc[nt].v, false, false);
    }
  }

  // ---- deterministic block merge across the 8 j-segments ----
  if (lane < 16) { marr[wv][lane] = m_run; larr[wv][lane] = l_run; }
  __syncthreads();
  float m_g = marr[0][q];
#pragma unroll
  for (int w = 1; w < 8; w++) m_g = fmaxf(m_g, marr[w][q]);
  float l_g = 0.f;
#pragma unroll
  for (int w = 0; w < 8; w++) l_g += larr[w][q] * __expf(marr[w][q] - m_g);
  float fw = __expf(m_run - m_g);
#pragma unroll
  for (int nt = 0; nt < 8; nt++)
#pragma unroll
    for (int v = 0; v < 8; v++) acc[nt].e[v] *= fw;

  // tree sum: element index = feature*16 + query
#define RIDX(nt, v) (((nt) * 16 + ((lane < 16) ? (v) : (v) + 8)) * 16 + q)
  if (wv >= 4)
#pragma unroll
    for (int nt = 0; nt < 8; nt++)
#pragma unroll
      for (int v = 0; v < 8; v++) redbuf[wv - 4][RIDX(nt, v)] = acc[nt].e[v];
  __syncthreads();
  if (wv < 4)
#pragma unroll
    for (int nt = 0; nt < 8; nt++)
#pragma unroll
      for (int v = 0; v < 8; v++) acc[nt].e[v] += redbuf[wv][RIDX(nt, v)];
  if (wv == 2 || wv == 3)
#pragma unroll
    for (int nt = 0; nt < 8; nt++)
#pragma unroll
      for (int v = 0; v < 8; v++) redbuf[wv][RIDX(nt, v)] = acc[nt].e[v];
  __syncthreads();
  if (wv < 2)
#pragma unroll
    for (int nt = 0; nt < 8; nt++)
#pragma unroll
      for (int v = 0; v < 8; v++) acc[nt].e[v] += redbuf[wv + 2][RIDX(nt, v)];
  if (wv == 1)
#pragma unroll
    for (int nt = 0; nt < 8; nt++)
#pragma unroll
      for (int v = 0; v < 8; v++) redbuf[1][RIDX(nt, v)] = acc[nt].e[v];
  __syncthreads();
  if (wv == 0) {                            // finalize: add, divide, stage tile
    float inv = 1.0f / l_g;
#pragma unroll
    for (int nt = 0; nt < 8; nt++)
#pragma unroll
      for (int v = 0; v < 8; v++)
        redbuf[0][RIDX(nt, v)] = (acc[nt].e[v] + redbuf[1][RIDX(nt, v)]) * inv;
  }
#undef RIDX
  __syncthreads();
  { // cooperative coalesced store: out[b, i0+q, :]
    int qq = t >> 4, c0 = (t & 15) * 8;
    float o0[8];
#pragma unroll
    for (int u = 0; u < 8; u++) o0[u] = redbuf[0][(c0 + u) * 16 + qq];
    float* op = out + (bN + i0 + qq) * (size_t)FOUT_ + c0;
    *(v4f*)op = *(v4f*)&o0[0];
    *(v4f*)(op + 4) = *(v4f*)&o0[4];
  }
}

// ---------------------------------------------------------------------------
extern "C" void kernel_launch(void* const* d_in, const int* in_sizes, int n_in,
                              void* d_out, int out_size, void* d_ws,
                              size_t ws_size, hipStream_t stream) {
  (void)in_sizes; (void)n_in; (void)out_size; (void)ws_size;
  const float* input = (const float*)d_in[0];
  const float* W     = (const float*)d_in[1];
  const float* a1    = (const float*)d_in[2];
  const float* a2    = (const float*)d_in[3];
  const int*   adj   = (const int*)d_in[4];
  // d_in[5] = adj_tree: unused by the reference forward
  const int*   d1p   = (const int*)d_in[6];
  const int*   d2p   = (const int*)d_in[7];
  float* out = (float*)d_out;

  char* ws = (char*)d_ws;
  _Float16* hT = (_Float16*)ws;                               // 2 MB
  _Float16* Wt = (_Float16*)(ws + (size_t)B_ * FOUT_ * N_ * 2); // 128 KB
  float* t1 = (float*)(ws + (size_t)B_ * FOUT_ * N_ * 2 + (size_t)FIN_ * FOUT_ * 2);
  float* t2 = t1 + (size_t)B_ * N_;

  gat_wt    <<<(FIN_ * FOUT_) / 256, 256, 0, stream>>>(W, Wt);
  gat_gemm_h<<<(B_ * N_) / 16,       256, 0, stream>>>(input, Wt, hT);
  gat_tvec  <<<(B_ * N_) / 256,      256, 0, stream>>>(hT, a1, a2, t1, t2);
  gat_attn  <<<(B_ * N_) / 16,       256, 0, stream>>>(hT, t1, t2, adj, d1p,
                                                       d2p, out);
}
